// MultiHeadedAttention_7026566496428
// MI455X (gfx1250) — matrix operand order repaired and checked by reading, not compiled, not run on hardware
//
#include <hip/hip_runtime.h>
#include <hip/hip_bf16.h>

// ---------------------------------------------------------------------------
// MultiHeadedAttention forward for MI455X (gfx1250), wave32 + WMMA f16.
//   B=2, S=2048, D=1024, H=16, DK=64
//   q/k/v = x @ Wqkv^T ; heads via reshape(B,H,S,DK) (faithful to reference);
//   flash-attention with int mask; out = attn @ Wout^T.
// ---------------------------------------------------------------------------

#define S_DIM 2048
#define D_DIM 1024
#define H_DIM 16
#define DK_DIM 64
#define BS_ROWS 4096   // B * S

typedef _Float16 v16h __attribute__((ext_vector_type(16)));
typedef _Float16 v8h  __attribute__((ext_vector_type(8)));
typedef _Float16 v4h  __attribute__((ext_vector_type(4)));
typedef float    v8f  __attribute__((ext_vector_type(8)));
typedef float    v4f  __attribute__((ext_vector_type(4)));
typedef int      gv4i __attribute__((vector_size(16)));   // matches builtin proto

// Async global->LDS copy (CDNA5, ASYNCcnt-tracked), guarded by __has_builtin.
#if defined(__gfx1250__) && __has_builtin(__builtin_amdgcn_global_load_async_to_lds_b128)
#define USE_ASYNC_LDS 1
#endif

__device__ __forceinline__ v16h cat8(v8h a, v8h b) {
    v16h r;
#pragma unroll
    for (int i = 0; i < 8; ++i) { r[i] = a[i]; r[i + 8] = b[i]; }
    return r;
}

__device__ __forceinline__ v8f wmma_f16(v16h a, v16h b, v8f c) {
    return __builtin_amdgcn_wmma_f32_16x16x32_f16(false, a, false, b,
                                                  (short)0, c, false, false);
}

// Head-row pointer for the reference's reshape(B,S,D)->(B,H,S,DK):
//   elem(b,h,s,dk) lives at row b*S + h*128 + s/16, col (s%16)*64 + dk.
__device__ __forceinline__ const _Float16* head_row(const _Float16* base,
                                                    int b, int h, int s) {
    return base + (size_t)(b * S_DIM + h * 128 + (s >> 4)) * D_DIM +
           (s & 15) * DK_DIM;
}

// ---------------------------------------------------------------------------
// GEMM: C(4096,1024) = A(4096,1024 f32) @ W(1024,1024 f32)^T, OutT = f16|f32.
// 128x128 block tile, BK=64; fp32 tiles register-prefetched one step ahead,
// converted to f16 and stored to LDS as packed b64.
// 8 waves: 4x2 grid, each wave 32x64 (2x4 WMMA tiles).
// ---------------------------------------------------------------------------
template <typename OutT>
__global__ __launch_bounds__(256) void gemm_xwt_kernel(const float* __restrict__ A,
                                                       const float* __restrict__ W,
                                                       OutT* __restrict__ C) {
    __shared__ __align__(128) _Float16 As[128][64];
    __shared__ __align__(128) _Float16 Bs[128][64];

    const int tid  = threadIdx.x;
    const int lane = tid & 31;
    const int wid  = tid >> 5;
    const int nB   = blockIdx.x * 128;   // col tile (over N = D)
    const int mB   = blockIdx.y * 128;   // row tile (over M = B*S)

    const int wm = wid >> 1;             // 0..3 -> 32-row band
    const int wn = wid & 1;              // 0..1 -> 64-col band
    const int mW = wm * 32;
    const int nW = wn * 64;

    const int mlane = lane & 15;
    const int half8 = (lane >> 4) * 8;   // 0 or 8

    // Staging role: 8 float4 chunks per thread per tile.
    int srow[8], scol[8];
#pragma unroll
    for (int j = 0; j < 8; ++j) {
        int idx = tid + 256 * j;          // float4 slot 0..2047
        srow[j] = idx >> 4;               // 16 float4 per row
        scol[j] = (idx & 15) << 2;
    }

    v8f acc[2][4];
#pragma unroll
    for (int mi = 0; mi < 2; ++mi)
#pragma unroll
        for (int ni = 0; ni < 4; ++ni) acc[mi][ni] = (v8f)0.0f;

    // Prologue: fetch tile kt=0 into registers.
    v4f ra[8], rw[8];
#pragma unroll
    for (int j = 0; j < 8; ++j) {
        ra[j] = *(const v4f*)(A + (size_t)(mB + srow[j]) * D_DIM + scol[j]);
        rw[j] = *(const v4f*)(W + (size_t)(nB + srow[j]) * D_DIM + scol[j]);
    }

    for (int kt = 0; kt < D_DIM; kt += 64) {
        // Commit prefetched registers to LDS (packed f16, b64 stores).
#pragma unroll
        for (int j = 0; j < 8; ++j) {
            v4h ha, hw;
#pragma unroll
            for (int e = 0; e < 4; ++e) {
                ha[e] = (_Float16)ra[j][e];
                hw[e] = (_Float16)rw[j][e];
            }
            *(v4h*)&As[srow[j]][scol[j]] = ha;
            *(v4h*)&Bs[srow[j]][scol[j]] = hw;
        }
        __syncthreads();

        // Register-prefetch next tile while WMMAs run on the current one.
        if (kt + 64 < D_DIM) {
#pragma unroll
            for (int j = 0; j < 8; ++j) {
                ra[j] = *(const v4f*)(A + (size_t)(mB + srow[j]) * D_DIM +
                                      kt + 64 + scol[j]);
                rw[j] = *(const v4f*)(W + (size_t)(nB + srow[j]) * D_DIM +
                                      kt + 64 + scol[j]);
            }
        }

#pragma unroll
        for (int ks = 0; ks < 64; ks += 32) {
            v16h af[2], bf[4];
#pragma unroll
            for (int mi = 0; mi < 2; ++mi) {
                const _Float16* r = &As[mW + mi * 16 + mlane][0];
                af[mi] = cat8(*(const v8h*)(r + ks + half8),
                              *(const v8h*)(r + ks + 16 + half8));
            }
#pragma unroll
            for (int ni = 0; ni < 4; ++ni)
                bf[ni] = cat8(*(const v8h*)&Bs[nW + ni * 16 + mlane][ks + half8],
                              *(const v8h*)&Bs[nW + ni * 16 + mlane][ks + 16 + half8]);
#pragma unroll
            for (int mi = 0; mi < 2; ++mi)
#pragma unroll
                for (int ni = 0; ni < 4; ++ni)
                    acc[mi][ni] = wmma_f16(af[mi], bf[ni], acc[mi][ni]);
        }
        __syncthreads();   // protect LDS before next commit
    }

    // Epilogue: C layout = VGPR r -> row r + 8*(lane>=16), lane%16 -> col.
#pragma unroll
    for (int mi = 0; mi < 2; ++mi)
#pragma unroll
        for (int ni = 0; ni < 4; ++ni)
#pragma unroll
            for (int r = 0; r < 8; ++r) {
                int row = mB + mW + mi * 16 + r + half8;
                int col = nB + nW + ni * 16 + mlane;
                C[(size_t)row * D_DIM + col] = (OutT)acc[mi][ni][r];
            }
}

// ---------------------------------------------------------------------------
// Flash attention. Grid: B*H*(S/128) = 512 blocks of 256 threads (8 waves).
// Each wave owns 16 query rows; block stages 32-key K/V tiles in LDS.
// ---------------------------------------------------------------------------
__global__ __launch_bounds__(256) void flash_attn_kernel(
    const _Float16* __restrict__ Q, const _Float16* __restrict__ K,
    const _Float16* __restrict__ V, const int* __restrict__ mask,
    float* __restrict__ Oout) {
    __shared__ __align__(128) _Float16 Ks[32][64];   // [key][dk]
    __shared__ __align__(128) _Float16 Vt[64][32];   // [dk][key] (transposed)
    __shared__ __align__(128) _Float16 Ps[8][16][32];// per-wave P bounce

    const int tid  = threadIdx.x;
    const int lane = tid & 31;
    const int wid  = tid >> 5;
    const int bx   = blockIdx.x;
    const int sT   = bx & 15;
    const int h    = (bx >> 4) & 15;
    const int b    = bx >> 8;
    const int sW   = sT * 128 + wid * 16;   // this wave's first query row

    const int nlan  = lane & 15;
    const int half  = lane >> 4;            // 0 or 1
    const int half8 = half * 8;

    // Load Q fragments once (A-matrix layout); fold 1/sqrt(DK)=0.125 in f16
    // (power of two -> exact), so scores need no post-scale.
    v16h qf[2];
    {
        const _Float16* qrow = head_row(Q, b, h, sW + nlan);
#pragma unroll
        for (int ks = 0; ks < 2; ++ks) {
            const _Float16* p = qrow + ks * 32;
            v16h t = cat8(*(const v8h*)(p + half8),
                          *(const v8h*)(p + 16 + half8));
#pragma unroll
            for (int e = 0; e < 16; ++e) t[e] = t[e] * (_Float16)0.125f;
            qf[ks] = t;
        }
    }

    float mrow[8], lrow[8];
    v8f o[4];
#pragma unroll
    for (int i = 0; i < 8; ++i) { mrow[i] = -3.0e38f; lrow[i] = 0.0f; }
#pragma unroll
    for (int nt = 0; nt < 4; ++nt) o[nt] = (v8f)0.0f;

    const int keyid = tid >> 3;        // 0..31, staging role
    const int dkb   = (tid & 7) * 8;   // 8-half chunk

    for (int kb = 0; kb < S_DIM; kb += 32) {
        __syncthreads();
        {   // stage K tile [32][64] and V tile transposed [64][32]
            int skey = kb + keyid;
            const _Float16* ksrc = head_row(K, b, h, skey) + dkb;
            _Float16*       kdst = &Ks[keyid][dkb];
#if defined(USE_ASYNC_LDS)
            __builtin_amdgcn_global_load_async_to_lds_b128(
                (__attribute__((address_space(1))) gv4i*)(void*)(_Float16*)ksrc,
                (__attribute__((address_space(3))) gv4i*)(void*)kdst, 0, 0);
#else
            *(v8h*)kdst = *(const v8h*)ksrc;
#endif
            v8h vv = *(const v8h*)(head_row(V, b, h, skey) + dkb);
#pragma unroll
            for (int j = 0; j < 8; ++j) Vt[dkb + j][keyid] = vv[j];
            if (kb + 32 < S_DIM) {     // global_prefetch_b8 next tiles
                __builtin_prefetch(head_row(K, b, h, kb + 32 + keyid) + dkb, 0, 1);
                __builtin_prefetch(head_row(V, b, h, kb + 32 + keyid) + dkb, 0, 1);
            }
        }
#if defined(USE_ASYNC_LDS)
#if __has_builtin(__builtin_amdgcn_s_wait_asynccnt)
        __builtin_amdgcn_s_wait_asynccnt(0);
#else
        asm volatile("s_wait_asynccnt 0x0" ::: "memory");
#endif
#endif
        __syncthreads();

        // scores: 2 tiles of 16x16, K-reduction over DK=64 in 2 WMMA steps.
        v8f sc[2];
        sc[0] = (v8f)0.0f; sc[1] = (v8f)0.0f;
#pragma unroll
        for (int t = 0; t < 2; ++t)
#pragma unroll
            for (int ks = 0; ks < 2; ++ks) {
                v16h bf = cat8(*(const v8h*)&Ks[t * 16 + nlan][ks * 32 + half8],
                               *(const v8h*)&Ks[t * 16 + nlan][ks * 32 + 16 + half8]);
                sc[t] = wmma_f16(qf[ks], bf, sc[t]);
            }

        // mask + online softmax (rows live in VGPR index, cols in lanes).
        const int* mbase = mask + (size_t)b * S_DIM * S_DIM +
                           (size_t)(sW + half8) * S_DIM + kb + nlan;
        float p0[8], p1[8], arow[8];
#pragma unroll
        for (int i = 0; i < 8; ++i) {
            float v0 = sc[0][i];
            float v1 = sc[1][i];
            if (mbase[(size_t)i * S_DIM] == 0)      v0 = -1.0e9f;
            if (mbase[(size_t)i * S_DIM + 16] == 0) v1 = -1.0e9f;
            float t = fmaxf(v0, v1);
#pragma unroll
            for (int off = 1; off < 16; off <<= 1)
                t = fmaxf(t, __shfl_xor(t, off, 32));
            float mnew = fmaxf(mrow[i], t);
            v0 = __expf(v0 - mnew);
            v1 = __expf(v1 - mnew);
            float rs = v0 + v1;
#pragma unroll
            for (int off = 1; off < 16; off <<= 1)
                rs += __shfl_xor(rs, off, 32);
            float al = __expf(mrow[i] - mnew);
            lrow[i] = lrow[i] * al + rs;
            mrow[i] = mnew;
            arow[i] = al;
            p0[i] = v0; p1[i] = v1;
        }
#pragma unroll
        for (int nt = 0; nt < 4; ++nt)
#pragma unroll
            for (int i = 0; i < 8; ++i) o[nt][i] *= arow[i];

        // Bounce P (C layout) through LDS to rebuild as WMMA A fragment.
        _Float16* pw = &Ps[wid][0][0];
#pragma unroll
        for (int i = 0; i < 8; ++i) {
            pw[(i + half8) * 32 + nlan]      = (_Float16)p0[i];
            pw[(i + half8) * 32 + 16 + nlan] = (_Float16)p1[i];
        }
        v16h pf = cat8(*(const v8h*)&Ps[wid][nlan][half8],
                       *(const v8h*)&Ps[wid][nlan][16 + half8]);

        // o += P @ V  (B fragments contiguous thanks to transposed Vt)
#pragma unroll
        for (int nt = 0; nt < 4; ++nt) {
            v16h vf = cat8(*(const v8h*)&Vt[nt * 16 + nlan][half8],
                           *(const v8h*)&Vt[nt * 16 + nlan][16 + half8]);
            o[nt] = wmma_f16(pf, vf, o[nt]);
        }
    }

    // Normalize and write x[b, s, h*64 + dk] (the transpose(0,2,1,3) reshape).
#pragma unroll
    for (int nt = 0; nt < 4; ++nt)
#pragma unroll
        for (int i = 0; i < 8; ++i) {
            int s   = sW + i + half8;
            int col = h * DK_DIM + nt * 16 + nlan;
            Oout[((size_t)(b * S_DIM + s)) * D_DIM + col] =
                o[nt][i] * (1.0f / lrow[i]);
        }
}

// ---------------------------------------------------------------------------
extern "C" void kernel_launch(void* const* d_in, const int* in_sizes, int n_in,
                              void* d_out, int out_size, void* d_ws, size_t ws_size,
                              hipStream_t stream) {
    (void)in_sizes; (void)n_in; (void)out_size; (void)ws_size;
    const float* query = (const float*)d_in[0];
    const float* keyi  = (const float*)d_in[1];
    const float* value = (const float*)d_in[2];
    const float* Wqkv  = (const float*)d_in[3];
    const float* Wout  = (const float*)d_in[4];
    const int*   mask  = (const int*)d_in[5];
    float*       out   = (float*)d_out;

    char* ws = (char*)d_ws;
    const size_t halfMat = (size_t)BS_ROWS * D_DIM * sizeof(_Float16); // 8 MB
    _Float16* Qf = (_Float16*)(ws);
    _Float16* Kf = (_Float16*)(ws + halfMat);
    _Float16* Vf = (_Float16*)(ws + 2 * halfMat);
    float*    Xf = (float*)(ws + 3 * halfMat);   // attention output, f32

    dim3 gemmGrid(D_DIM / 128, BS_ROWS / 128);   // (8, 32)
    dim3 blk(256);

    gemm_xwt_kernel<_Float16><<<gemmGrid, blk, 0, stream>>>(query, Wqkv, Qf);
    gemm_xwt_kernel<_Float16><<<gemmGrid, blk, 0, stream>>>(keyi,  Wqkv, Kf);
    gemm_xwt_kernel<_Float16><<<gemmGrid, blk, 0, stream>>>(value, Wqkv, Vf);

    flash_attn_kernel<<<dim3(2 * H_DIM * (S_DIM / 128)), blk, 0, stream>>>(
        Qf, Kf, Vf, mask, Xf);

    gemm_xwt_kernel<float><<<gemmGrid, blk, 0, stream>>>(Xf, Wout, out);
}
